// EncoderNum_40862318854489
// MI455X (gfx1250) — matrix operand order repaired and checked by reading, not compiled, not run on hardware
//
#include <hip/hip_runtime.h>
#include <hip/hip_bf16.h>

// ---------------------------------------------------------------------------
// EncoderNum: ragged gather + concat + masked segment-mean + pad, reformulated
// as a histogram GEMM:  out[r,:] = (counts[r,0:32] @ Emb[32,512]) * (1/len[r])
// with padding rows replaced by num_padding_hidden.
// Emb rows 0..9  = digit_emb (cols 0..255, zero elsewhere)
// Emb rows 10..28= pos_emb   (cols 256..511, zero elsewhere)
// f16 WMMA 16x16x32 with hi/lo residual split of Emb for ~fp32 accuracy.
// ---------------------------------------------------------------------------

typedef __attribute__((ext_vector_type(8)))  _Float16 v8h;
typedef __attribute__((ext_vector_type(16))) _Float16 v16h;
typedef __attribute__((ext_vector_type(8)))  float    v8f;

#define BB 64
#define NN 128
#define DD 19
#define HH 512
#define ROWS (BB * NN)          // 8192
#define KK 32                   // padded K (10 digit + 19 pos + 3 zero)

// ---- Kernel 1: build combined embedding table, f16 hi + f16 residual lo ----
__global__ __launch_bounds__(256) void build_emb(const float* __restrict__ dig,
                                                 const float* __restrict__ pos,
                                                 _Float16* __restrict__ hi,
                                                 _Float16* __restrict__ lo) {
    int i = blockIdx.x * blockDim.x + threadIdx.x;   // 0 .. 32*512-1
    if (i >= KK * HH) return;
    int k = i >> 9;          // row 0..31
    int c = i & (HH - 1);    // col 0..511
    float v = 0.0f;
    if (k < 10) {
        if (c < HH / 2) v = dig[k * (HH / 2) + c];
    } else if (k < 29) {
        if (c >= HH / 2) v = pos[(k - 10) * (HH / 2) + (c - HH / 2)];
    }
    _Float16 h = (_Float16)v;
    hi[i] = h;
    lo[i] = (_Float16)(v - (float)h);
}

// ---- Kernel 2: per-slot digit/pos histograms (exact small ints in f16) ----
__global__ __launch_bounds__(256) void build_counts(const int* __restrict__ nd,
                                                    const int* __restrict__ np,
                                                    const int* __restrict__ dl,
                                                    _Float16* __restrict__ cnt,
                                                    float* __restrict__ recip) {
    int r = blockIdx.x * blockDim.x + threadIdx.x;   // 0 .. 8191
    if (r >= ROWS) return;
    int len = dl[r];
    if (len < 1) len = 1;
    int dg[DD], ps[DD];
#pragma unroll
    for (int d = 0; d < DD; ++d) {
        dg[d] = nd[r * DD + d];
        ps[d] = np[r * DD + d];
    }
    recip[r] = 1.0f / (float)len;
    _Float16* o = cnt + r * KK;
#pragma unroll
    for (int k = 0; k < KK; ++k) {
        int s = 0;
        if (k < 10) {
#pragma unroll
            for (int d = 0; d < DD; ++d) s += (d < len && dg[d] == k) ? 1 : 0;
        } else if (k < 29) {
#pragma unroll
            for (int d = 0; d < DD; ++d) s += (d < len && ps[d] == (k - 10)) ? 1 : 0;
        }
        o[k] = (_Float16)s;
    }
}

// ---- Kernel 3: WMMA GEMM + scale + pad epilogue -----------------------------
// One wave per 16x16 output tile. 512 row-tiles x 32 col-tiles = 16384 waves.
__global__ __launch_bounds__(256) void wmma_main(const _Float16* __restrict__ embHi,
                                                 const _Float16* __restrict__ embLo,
                                                 const _Float16* __restrict__ cnt,
                                                 const float* __restrict__ recip,
                                                 const int* __restrict__ num_counts,
                                                 const float* __restrict__ pad_h,
                                                 float* __restrict__ out) {
    const int lane = threadIdx.x & 31;
    const int wave = threadIdx.x >> 5;
    const int tile = blockIdx.x * 8 + wave;      // 0..16383
    const int trow = tile >> 5;                  // 0..511 (row tile)
    const int tcol = tile & 31;                  // 0..31  (col tile)
    const int hi   = lane >> 4;                  // half-wave select
    const int l16  = lane & 15;

    // A operand: 16x32 f16 counts tile.
    // ISA 16-bit A layout: lane M = lane&15; lanes 0-15 hold K=0..7 (e0..7) and
    // K=16..23 (e8..15); lanes 16-31 hold K=8..15 and K=24..31.
    const _Float16* arow = cnt + (size_t)(trow * 16 + l16) * KK;
    v8h a0 = *(const v8h*)(arow + 8 * hi);        // K = 8*hi .. 8*hi+7
    v8h a1 = *(const v8h*)(arow + 16 + 8 * hi);   // K = 16+8*hi .. 16+8*hi+7
    v16h A;
#pragma unroll
    for (int i = 0; i < 8; ++i) { A[i] = a0[i]; A[i + 8] = a1[i]; }

    // B operand: 32x16 f16 emb tile. lane N = lane&15; elements follow A's K order
    // (e0..7: K=8*hi+e, e8..15: K=16+8*hi+(e-8)) -> strided column gather.
    const size_t boff = (size_t)tcol * 16 + l16;
    v16h Bh, Bl;
#pragma unroll
    for (int i = 0; i < 8; ++i) {
        Bh[i]     = embHi[boff + (size_t)(8 * hi + i) * HH];
        Bh[i + 8] = embHi[boff + (size_t)(16 + 8 * hi + i) * HH];
        Bl[i]     = embLo[boff + (size_t)(8 * hi + i) * HH];
        Bl[i + 8] = embLo[boff + (size_t)(16 + 8 * hi + i) * HH];
    }

    v8f acc = {};
    acc = __builtin_amdgcn_wmma_f32_16x16x32_f16(false, A, false, Bh,
                                                 (short)0, acc, false, false);
    acc = __builtin_amdgcn_wmma_f32_16x16x32_f16(false, A, false, Bl,
                                                 (short)0, acc, false, false);

    // Epilogue. C/D layout: lane N = lane&15; VGPR j -> row M = j + 8*hi.
    const int rbase = trow * 16 + 8 * hi;        // this lane's 8 consecutive rows
    v8f rc = *(const v8f*)(recip + rbase);       // 1/len per row
    const int b = (trow * 16) >> 7;              // batch (tile never crosses b)
    int c = num_counts[b];
    if (c < 1) c = 1;
    const int col = tcol * 16 + l16;
    const float pv = pad_h[col];
    float* orow = out + (size_t)rbase * HH + col;
#pragma unroll
    for (int j = 0; j < 8; ++j) {
        int n = (rbase + j) & (NN - 1);
        float v = (n < c) ? acc[j] * rc[j] : pv;
        orow[(size_t)j * HH] = v;
    }
}

extern "C" void kernel_launch(void* const* d_in, const int* in_sizes, int n_in,
                              void* d_out, int out_size, void* d_ws, size_t ws_size,
                              hipStream_t stream) {
    const float* pad_h = (const float*)d_in[0];   // [512]
    const float* dig   = (const float*)d_in[1];   // [10,256]
    const float* pos   = (const float*)d_in[2];   // [19,256]
    const int*   nd    = (const int*)d_in[3];     // [64,128,19]
    const int*   np    = (const int*)d_in[4];     // [64,128,19]
    const int*   dl    = (const int*)d_in[5];     // [64,128]
    const int*   nc    = (const int*)d_in[6];     // [64]
    float*       out   = (float*)d_out;           // [64,128,512]

    // Workspace layout (all regions fully rewritten every call):
    //   embHi : 32*512 f16   (32 KB)
    //   embLo : 32*512 f16   (32 KB)
    //   cnt   : 8192*32 f16  (512 KB)
    //   recip : 8192 f32     (32 KB)
    char* ws = (char*)d_ws;
    _Float16* embHi = (_Float16*)ws;
    _Float16* embLo = embHi + KK * HH;
    _Float16* cntp  = embLo + KK * HH;
    float*    recip = (float*)(cntp + (size_t)ROWS * KK);

    build_emb<<<(KK * HH + 255) / 256, 256, 0, stream>>>(dig, pos, embHi, embLo);
    build_counts<<<(ROWS + 255) / 256, 256, 0, stream>>>(nd, np, dl, cntp, recip);

    // 512 row-tiles * 32 col-tiles = 16384 tiles, 8 waves (tiles) per block.
    wmma_main<<<16384 / 8, 256, 0, stream>>>(embHi, embLo, cntp, recip, nc, pad_h, out);
}
